// mie_51281909514553
// MI455X (gfx1250) — matrix operand order repaired and checked by reading, not compiled, not run on hardware
//
#include <hip/hip_runtime.h>
#include <hip/hip_bf16.h>

typedef float  v8f  __attribute__((ext_vector_type(8)));
typedef __bf16 v8bf __attribute__((ext_vector_type(8)));
typedef __bf16 v16bf __attribute__((ext_vector_type(16)));

// ---------- helpers ----------
static __device__ __forceinline__ __bf16 f32_to_bf16(float f) {
    union { float f; unsigned u; } in; in.f = f;
    unsigned r = in.u + 0x7FFFu + ((in.u >> 16) & 1u);   // round-to-nearest-even
    union { unsigned short s; __bf16 b; } out;
    out.s = (unsigned short)(r >> 16);
    return out.b;
}

// low 32 bits of a flat shared-aperture address == wave-relative LDS byte address
static __device__ __forceinline__ unsigned lds_off(const void* p) {
    return (unsigned)(uintptr_t)p;
}

// async 16B global -> LDS copy (tracked by ASYNCcnt)
#define ASYNC_CP_B128(loff, gptr)                                              \
    asm volatile("global_load_async_to_lds_b128 %0, %1, off"                   \
                 :: "v"(loff), "v"(gptr) : "memory")

static __device__ __forceinline__ void wait_async0() {
    asm volatile("s_wait_asynccnt 0x0" ::: "memory");
}

// ---------- x (f32) -> bf16 ----------
__global__ __launch_bounds__(256) void cvt_f32_bf16(const float* __restrict__ src,
                                                    __bf16* __restrict__ dst, int n) {
    int i = blockIdx.x * 256 + threadIdx.x;
    if (i < n) dst[i] = f32_to_bf16(src[i]);
}

// ---------- merged, TRANSPOSED expert weights: WT[o*K + k] = sum_e wn_e * W[e][k][o] ----------
__global__ __launch_bounds__(256) void merge_wT(const float* __restrict__ W,   // [E,K,N]
                                                const float* __restrict__ wv,  // [8]
                                                __bf16* __restrict__ WT,       // [N,K]
                                                int K, int N) {
    __shared__ float tile[32][33];
    float s = 0.f;
#pragma unroll
    for (int e = 0; e < 8; ++e) s += wv[e];
    float wn[8];
#pragma unroll
    for (int e = 0; e < 8; ++e) wn[e] = wv[e] / s;

    const int k0 = blockIdx.y * 32;
    const int o0 = blockIdx.x * 32;
    const int tx = threadIdx.x;           // 0..31
    const int ty = threadIdx.y;           // 0..7

    for (int r = ty; r < 32; r += 8) {
        const int k = k0 + r;
        const int o = o0 + tx;
        float acc = 0.f;
#pragma unroll
        for (int e = 0; e < 8; ++e)
            acc += wn[e] * W[(size_t)e * K * N + (size_t)k * N + o];
        tile[r][tx] = acc;
    }
    __syncthreads();
    for (int r = ty; r < 32; r += 8) {
        const int o = o0 + r;
        const int k = k0 + tx;
        WT[(size_t)o * K + k] = f32_to_bf16(tile[tx][r]);
    }
}

// ---------- merged bias: bm[o] = sum_e wn_e * b[e][o] ----------
__global__ __launch_bounds__(256) void merge_b(const float* __restrict__ b,   // [E,N]
                                               const float* __restrict__ wv,
                                               float* __restrict__ bm, int N) {
    int o = blockIdx.x * 256 + threadIdx.x;
    if (o >= N) return;
    float s = 0.f;
#pragma unroll
    for (int e = 0; e < 8; ++e) s += wv[e];
    float acc = 0.f;
#pragma unroll
    for (int e = 0; e < 8; ++e) acc += (wv[e] / s) * b[(size_t)e * N + o];
    bm[o] = acc;
}

// ---------- WMMA GEMM + bias + PReLU, async-LDS staged, double buffered ----------
// A: [M,K] bf16 row-major.  WT: [N,K] bf16 (Wm transposed).
// Block: 256 threads (8 wave32), tile 128(M) x 256(N); wave tile 64x64 = 4x4 wmma.
// grid = (N/256, M/128).  K, N multiples of 32 / 256.
template <bool FINAL>
__global__ __launch_bounds__(256) void gemm_bias_prelu(
    const __bf16* __restrict__ A, const __bf16* __restrict__ WT,
    const float* __restrict__ bias, const float* __restrict__ alpha,
    void* __restrict__ outv, int K, int N) {

    __shared__ __align__(16) __bf16 As[2][128][32];   // 2 x 8 KB
    __shared__ __align__(16) __bf16 Bs[2][256][32];   // 2 x 16 KB

    const int tid  = threadIdx.x;
    const int lane = tid & 31;
    const int wave = tid >> 5;
    const int wm   = wave & 1;       // 0..1  (M direction)
    const int wnv  = wave >> 1;      // 0..3  (N direction)
    const int ml   = lane & 15;      // row/col within 16x16 tile
    const int h    = lane >> 4;      // half-wave select

    const int mblk = blockIdx.y * 128;
    const int nblk = blockIdx.x * 256;

    // ---- async copy mapping: 16B chunks; row = chunk>>2, col-chunk = chunk&3 ----
    const int crow = tid >> 2;               // 0..63
    const int ccol = (tid & 3) * 8;          // element offset 0,8,16,24
    const __bf16* agp = A  + (size_t)(mblk + crow) * K + ccol;   // rows crow, crow+64
    const __bf16* bgp = WT + (size_t)(nblk + crow) * K + ccol;   // rows crow + 64*j
    const unsigned aoff = lds_off(&As[0][crow][ccol]);           // +64*64 for 2nd half
    const unsigned boff = lds_off(&Bs[0][crow][ccol]);           // +j*4096 per 64 rows
    const unsigned ABUF = 128 * 32 * 2;      // 8192 B per A buffer
    const unsigned BBUF = 256 * 32 * 2;      // 16384 B per B buffer

    auto issue_tile = [&](int ib, int kk) {
        const unsigned ab  = aoff + (unsigned)ib * ABUF;
        const unsigned bbv = boff + (unsigned)ib * BBUF;
        ASYNC_CP_B128(ab,              agp + kk);
        ASYNC_CP_B128(ab + 64u * 64u,  agp + (size_t)64 * K + kk);
#pragma unroll
        for (int j = 0; j < 4; ++j) {
            ASYNC_CP_B128(bbv + (unsigned)j * 4096u, bgp + (size_t)(j * 64) * K + kk);
        }
    };

    v8f acc[4][4];
#pragma unroll
    for (int r = 0; r < 4; ++r)
#pragma unroll
        for (int c = 0; c < 4; ++c) acc[r][c] = {};

    issue_tile(0, 0);

    int ib = 0;
    for (int kk = 0; kk < K; kk += 32, ib ^= 1) {
        wait_async0();          // this wave's async copies into buf[ib] done
        __syncthreads();        // everyone's copies done, previous reads retired

        if (kk + 32 < K) issue_tile(ib ^ 1, kk + 32);

        // A fragments (16x32 bf16, contiguous k order, not ISA table order): lane ml = row,
        // elems 0..15 = K h*16..h*16+15 (contiguous, matches B)  (two 16B DS reads)
        v16bf afr[4];
#pragma unroll
        for (int r = 0; r < 4; ++r) {
            const __bf16* p = &As[ib][wm * 64 + r * 16 + ml][0];
            v8bf lo = *(const v8bf*)(p + h * 16);
            v8bf hi = *(const v8bf*)(p + h * 16 + 8);
            afr[r] = __builtin_shufflevector(lo, hi, 0, 1, 2, 3, 4, 5, 6, 7,
                                             8, 9, 10, 11, 12, 13, 14, 15);
        }
        // B fragments: lane ml = column; 16 consecutive K from h*16 (one 32B DS read).
        // Hoisted into an array so all 4 DS loads are in flight before the WMMA block.
        v16bf bfr[4];
#pragma unroll
        for (int c = 0; c < 4; ++c) {
            const __bf16* q = &Bs[ib][wnv * 64 + c * 16 + ml][h * 16];
            bfr[c] = *(const v16bf*)q;
        }
#pragma unroll
        for (int c = 0; c < 4; ++c) {
#pragma unroll
            for (int r = 0; r < 4; ++r) {
                acc[r][c] = __builtin_amdgcn_wmma_f32_16x16x32_bf16(
                    false, afr[r], false, bfr[c], (short)0, acc[r][c], false, false);
            }
        }
    }

    // Epilogue: C/D layout -> element v of lane = C[h*8 + v][ml] within tile.
#pragma unroll
    for (int c = 0; c < 4; ++c) {
        const int col = nblk + wnv * 64 + c * 16 + ml;
        const float bcol = bias[col];
        const float acol = alpha[col];
#pragma unroll
        for (int r = 0; r < 4; ++r) {
#pragma unroll
            for (int v = 0; v < 8; ++v) {
                const int row = mblk + wm * 64 + r * 16 + h * 8 + v;
                float val = acc[r][c][v] + bcol;
                val = (val >= 0.f) ? val : acol * val;
                if (FINAL)
                    ((float*)outv)[(size_t)row * N + col] = val;
                else
                    ((__bf16*)outv)[(size_t)row * N + col] = f32_to_bf16(val);
            }
        }
    }
}

// ---------- launch ----------
extern "C" void kernel_launch(void* const* d_in, const int* in_sizes, int n_in,
                              void* d_out, int out_size, void* d_ws, size_t ws_size,
                              hipStream_t stream) {
    (void)in_sizes; (void)n_in; (void)out_size; (void)ws_size;

    constexpr int B = 8192;
    constexpr int D0 = 512, D1 = 1024, D2 = 1024, D3 = 512;

    const float* x  = (const float*)d_in[0];
    const float* wv = (const float*)d_in[1];
    const float* W[3]  = { (const float*)d_in[2], (const float*)d_in[5], (const float*)d_in[8] };
    const float* bb[3] = { (const float*)d_in[3], (const float*)d_in[6], (const float*)d_in[9] };
    const float* aa[3] = { (const float*)d_in[4], (const float*)d_in[7], (const float*)d_in[10] };

    // workspace layout (bytes)
    char* ws = (char*)d_ws;
    constexpr size_t OFF_XB   = 0;
    constexpr size_t OFF_ACT1 = OFF_XB   + (size_t)B * D0 * 2;
    constexpr size_t OFF_ACT2 = OFF_ACT1 + (size_t)B * D1 * 2;
    constexpr size_t OFF_WT0  = OFF_ACT2 + (size_t)B * D2 * 2;
    constexpr size_t OFF_WT1  = OFF_WT0  + (size_t)D0 * D1 * 2;
    constexpr size_t OFF_WT2  = OFF_WT1  + (size_t)D1 * D2 * 2;
    constexpr size_t OFF_BM0  = OFF_WT2  + (size_t)D2 * D3 * 2;
    constexpr size_t OFF_BM1  = OFF_BM0  + (size_t)D1 * 4;
    constexpr size_t OFF_BM2  = OFF_BM1  + (size_t)D2 * 4;

    __bf16* xb   = (__bf16*)(ws + OFF_XB);
    __bf16* act1 = (__bf16*)(ws + OFF_ACT1);
    __bf16* act2 = (__bf16*)(ws + OFF_ACT2);
    __bf16* WT0  = (__bf16*)(ws + OFF_WT0);
    __bf16* WT1  = (__bf16*)(ws + OFF_WT1);
    __bf16* WT2  = (__bf16*)(ws + OFF_WT2);
    float*  bm0  = (float*)(ws + OFF_BM0);
    float*  bm1  = (float*)(ws + OFF_BM1);
    float*  bm2  = (float*)(ws + OFF_BM2);

    // 1) x -> bf16
    {
        int n = B * D0;
        cvt_f32_bf16<<<(n + 255) / 256, 256, 0, stream>>>(x, xb, n);
    }
    // 2) merge weights (transposed bf16) + biases
    dim3 tb(32, 8);
    merge_wT<<<dim3(D1 / 32, D0 / 32), tb, 0, stream>>>(W[0], wv, WT0, D0, D1);
    merge_wT<<<dim3(D2 / 32, D1 / 32), tb, 0, stream>>>(W[1], wv, WT1, D1, D2);
    merge_wT<<<dim3(D3 / 32, D2 / 32), tb, 0, stream>>>(W[2], wv, WT2, D2, D3);
    merge_b<<<(D1 + 255) / 256, 256, 0, stream>>>(bb[0], wv, bm0, D1);
    merge_b<<<(D2 + 255) / 256, 256, 0, stream>>>(bb[1], wv, bm1, D2);
    merge_b<<<(D3 + 255) / 256, 256, 0, stream>>>(bb[2], wv, bm2, D3);

    // 3) three WMMA GEMMs with fused bias + PReLU
    gemm_bias_prelu<false><<<dim3(D1 / 256, B / 128), 256, 0, stream>>>(xb,   WT0, bm0, aa[0], act1, D0, D1);
    gemm_bias_prelu<false><<<dim3(D2 / 256, B / 128), 256, 0, stream>>>(act1, WT1, bm1, aa[1], act2, D1, D2);
    gemm_bias_prelu<true ><<<dim3(D3 / 256, B / 128), 256, 0, stream>>>(act2, WT2, bm2, aa[2], d_out, D2, D3);
}
